// VGGBlockInput_83708912599750
// MI455X (gfx1250) — matrix operand order repaired and checked by reading, not compiled, not run on hardware
//
#include <hip/hip_runtime.h>
#include <hip/hip_bf16.h>

// ---------- types ----------
typedef __attribute__((ext_vector_type(16))) __bf16 v16bf;
typedef __attribute__((ext_vector_type(8)))  __bf16 v8bf;
typedef __attribute__((ext_vector_type(8)))  float  v8f;

__device__ __forceinline__ unsigned short f2bfbits(float f) {
    unsigned int u = __float_as_uint(f);
    unsigned int r = u + 0x7FFFu + ((u >> 16) & 1u);   // round-to-nearest-even
    return (unsigned short)(r >> 16);
}
__device__ __forceinline__ float bf2f(unsigned short h) {
    return __uint_as_float(((unsigned int)h) << 16);
}

// ---------- problem constants ----------
#define MTOT (16L * 256L * 256L)      // flattened N*H*W
#define NFRAG_ELEMS 512               // one 32x16 bf16 B fragment
#define NFRAGS (9 * 2 * 4)            // taps * kchunks * ntiles
#define COLSTRIDE 72                  // bf16 elems per (row,col) LDS slot (64 + pad)

#define ACT_ELEMS (3 * 130 * COLSTRIDE)
#define ACT_BYTES (ACT_ELEMS * 2)                 // 56160
#define WP_ELEMS  (NFRAGS * NFRAG_ELEMS)          // 36864
#define WP_BYTES  (WP_ELEMS * 2)                  // 73728

// gfx1250 async global->LDS path (guarded; falls back to VGPR copies)
#if defined(__has_builtin)
#if __has_builtin(__builtin_amdgcn_global_load_async_to_lds_b128) && \
    __has_builtin(__builtin_amdgcn_s_wait_asynccnt)
#define USE_ASYNC_LDS 1
#endif
#endif

// builtin expects: (int4* AS1 gsrc, int4* AS3 ldst, int imm_offset, int cpol)
typedef int int4v __attribute__((vector_size(16)));
typedef __attribute__((address_space(1))) int4v* gv4ptr;
typedef __attribute__((address_space(3))) int4v* lv4ptr;

// =====================================================================
// Weight packing: OIHW fp32 [64][64][3][3] -> per-lane B-fragment bf16.
// Fragment f = (t*2+q)*4 + j ; element e = lane*16 + i
//   lane<16 : N = j*16+lane,    K = i
//   lane>=16: N = j*16+lane-16, K = 16+i
//   input channel cin = q*32 + K ; tap t = ky*3+kx
// =====================================================================
__global__ __launch_bounds__(256) void pack_w_kernel(const float* __restrict__ w,
                                                     unsigned short* __restrict__ wp) {
    int id = blockIdx.x * 256 + threadIdx.x;
    if (id >= NFRAGS * NFRAG_ELEMS) return;
    int e    = id & 511;
    int f    = id >> 9;
    int lane = e >> 4;
    int i    = e & 15;
    int t    = f >> 3;
    int q    = (f >> 2) & 1;
    int j    = f & 3;
    int nOut = j * 16 + (lane & 15);
    int kk   = (lane < 16) ? i : (16 + i);
    int cin  = q * 32 + kk;
    wp[id]   = f2bfbits(w[(nOut * 64 + cin) * 9 + t]);
}

// =====================================================================
// Conv1: direct 3x3, Cin=3 -> 64, + bias + PReLU, bf16 channels-last out
// =====================================================================
__global__ __launch_bounds__(256) void conv1_kernel(const float* __restrict__ x,
                                                    const float* __restrict__ w1,
                                                    const float* __restrict__ b1,
                                                    const float* __restrict__ prelu_a,
                                                    unsigned short* __restrict__ x0) {
    int tid = threadIdx.x;
    long s  = (long)blockIdx.x * 4 + (tid >> 6);
    int c   = tid & 63;
    int n   = (int)(s >> 16);
    int h   = (int)((s >> 8) & 255);
    int w   = (int)(s & 255);
    float acc = b1[c];
    const float* wc = w1 + c * 27;
#pragma unroll
    for (int ci = 0; ci < 3; ++ci)
#pragma unroll
        for (int dy = 0; dy < 3; ++dy) {
            int hin = h + dy - 1;
#pragma unroll
            for (int dx = 0; dx < 3; ++dx) {
                int win = w + dx - 1;
                float xv = 0.f;
                if ((unsigned)hin < 256u && (unsigned)win < 256u)
                    xv = x[((long)n * 3 + ci) * 65536L + hin * 256 + win];
                acc = fmaf(xv, wc[ci * 9 + dy * 3 + dx], acc);
            }
        }
    float a = prelu_a[0];
    acc = acc >= 0.f ? acc : a * acc;
    x0[s * 64 + c] = f2bfbits(acc);
}

// =====================================================================
// Implicit-GEMM 3x3 conv (64->64) via v_wmma_f32_16x16x32_bf16.
// Block = 256 threads = 8 wave32, one 128-wide row strip per block.
// LDS: activations (3 rows x 130 halo cols x 64ch bf16, zero-padded)
//      + full packed weight set (72KB, async-DMA'd once per block).
// Each wave: 16 spatial x 64 outch tile, K = 9 taps x 2 chunks -> 72 WMMA.
// FUSE=true : out = prelu(max(resid, conv+bias)) -> bf16 channels-last
// FUSE=false: out = conv+bias -> fp32 NCHW via LDS-transposed coalesced
//             float4 stores.
// =====================================================================
template <bool FUSE>
__global__ __launch_bounds__(256, 1) void conv_wmma_kernel(
        const unsigned short* __restrict__ act,    // [M][64] bf16 channels-last
        const unsigned short* __restrict__ wpack,  // packed B fragments
        const float* __restrict__ bias,            // [64]
        const unsigned short* __restrict__ resid,  // x0 (FUSE only)
        const float* __restrict__ prelu_a,
        unsigned short* __restrict__ outBf,        // (FUSE only)
        float* __restrict__ outF32) {              // (!FUSE only)
    __shared__ __align__(16) unsigned char smem[ACT_BYTES + WP_BYTES];
    unsigned short* lds_act = (unsigned short*)smem;
    unsigned short* lds_w   = (unsigned short*)(smem + ACT_BYTES);

    const int tid = threadIdx.x;
    const long s0 = (long)blockIdx.x * 128;
    const int n   = (int)(s0 >> 16);
    const int h   = (int)((s0 >> 8) & 255);
    const int w0  = (int)(s0 & 255);   // 0 or 128

    // ---- stage packed weights into LDS (async DMA if available) ----
#ifdef USE_ASYNC_LDS
    {
        gv4ptr gsrc = (gv4ptr)wpack;
        lv4ptr ldst = (lv4ptr)(smem + ACT_BYTES);
        for (int i = tid; i < WP_BYTES / 16; i += 256)
            __builtin_amdgcn_global_load_async_to_lds_b128(gsrc + i, ldst + i, 0, 0);
    }
#else
    for (int idx = tid; idx < WP_BYTES / 16; idx += 256) {
        uint4 v = *(const uint4*)((const char*)wpack + idx * 16);
        *(uint4*)(smem + ACT_BYTES + idx * 16) = v;
    }
#endif

    // ---- stage activations (zero-padded halos) into LDS ----
    for (int idx = tid; idx < 3 * 130 * 8; idx += 256) {
        int dy  = idx / (130 * 8);
        int rem = idx - dy * 130 * 8;
        int col = rem >> 3;
        int cs  = (rem & 7) * 8;                 // channel start (8 bf16 = 16B)
        int hin = h + dy - 1;
        int win = w0 - 1 + col;
        uint4 val = make_uint4(0u, 0u, 0u, 0u);
        if ((unsigned)hin < 256u && (unsigned)win < 256u) {
            long sIn = (long)n * 65536L + hin * 256 + win;
            val = *(const uint4*)(act + sIn * 64 + cs);
        }
        *(uint4*)(lds_act + (dy * 130 + col) * COLSTRIDE + cs) = val;
    }

#ifdef USE_ASYNC_LDS
    __builtin_amdgcn_s_wait_asynccnt(0);
#endif
    __syncthreads();

    const int lane = tid & 31;
    const int wv   = tid >> 5;
    const int l15  = lane & 15;
    const int hsel = lane >> 4;      // 0 or 1
    const unsigned short* wl = lds_w + lane * 16;   // per-lane fragment base

    v8f acc0 = {}, acc1 = {}, acc2 = {}, acc3 = {};

    for (int t = 0; t < 9; ++t) {
        const int ky = t / 3, kx = t % 3;
#pragma unroll
        for (int q = 0; q < 2; ++q) {
            // A fragment: lane<16 -> K=q*32+{0..15}; lane>=16 -> {16..31}
            const unsigned short* ap =
                lds_act + (ky * 130 + wv * 16 + l15 + kx) * COLSTRIDE + q * 32 + hsel * 16;
            v8bf lo = *(const v8bf*)ap;          // ds_load_b128
            v8bf hi = *(const v8bf*)(ap + 8);    // ds_load_b128
            v16bf a = __builtin_shufflevector(lo, hi, 0, 1, 2, 3, 4, 5, 6, 7, 8, 9,
                                              10, 11, 12, 13, 14, 15);

            const int fb = ((t * 2 + q) * 4) * NFRAG_ELEMS;
            v16bf b0 = *(const v16bf*)(wl + fb);
            v16bf b1 = *(const v16bf*)(wl + fb + 512);
            v16bf b2 = *(const v16bf*)(wl + fb + 1024);
            v16bf b3 = *(const v16bf*)(wl + fb + 1536);

            acc0 = __builtin_amdgcn_wmma_f32_16x16x32_bf16(false, a, false, b0,
                                                           (short)0, acc0, false, false);
            acc1 = __builtin_amdgcn_wmma_f32_16x16x32_bf16(false, a, false, b1,
                                                           (short)0, acc1, false, false);
            acc2 = __builtin_amdgcn_wmma_f32_16x16x32_bf16(false, a, false, b2,
                                                           (short)0, acc2, false, false);
            acc3 = __builtin_amdgcn_wmma_f32_16x16x32_bf16(false, a, false, b3,
                                                           (short)0, acc3, false, false);
        }
    }

    const long sBase = s0 + wv * 16;

    if (FUSE) {
        // ---- fused epilogue: bias + max(x0, .) + PReLU -> bf16 ----
        const float aP = prelu_a[0];
#define EPILOG_F(JJ, ACC)                                                        \
        {                                                                        \
            const int c = (JJ)*16 + l15;                                         \
            const float bc = bias[c];                                            \
            _Pragma("unroll") for (int r = 0; r < 8; ++r) {                      \
                const int m = r + hsel * 8;                                      \
                const long s = sBase + m;                                        \
                float v = (ACC)[r] + bc;                                         \
                float x0v = bf2f(resid[s * 64 + c]);                             \
                float mx = fmaxf(x0v, v);                                        \
                mx = mx >= 0.f ? mx : aP * mx;                                   \
                outBf[s * 64 + c] = f2bfbits(mx);                                \
            }                                                                    \
        }
        EPILOG_F(0, acc0)
        EPILOG_F(1, acc1)
        EPILOG_F(2, acc2)
        EPILOG_F(3, acc3)
#undef EPILOG_F
    } else {
        // ---- conv3 epilogue: bias add, LDS transpose, coalesced f32 NCHW ----
        __syncthreads();                     // done with act/weight LDS
        float* lds_t = (float*)smem;         // [64][132] f32 (33792 B)
#define CST 132
#define EPILOG_T(JJ, ACC)                                                        \
        {                                                                        \
            const int c = (JJ)*16 + l15;                                         \
            const float bc = bias[c];                                            \
            _Pragma("unroll") for (int r = 0; r < 8; ++r) {                      \
                const int m = r + hsel * 8;                                      \
                lds_t[c * CST + wv * 16 + m] = (ACC)[r] + bc;                    \
            }                                                                    \
        }
        EPILOG_T(0, acc0)
        EPILOG_T(1, acc1)
        EPILOG_T(2, acc2)
        EPILOG_T(3, acc3)
#undef EPILOG_T
        __syncthreads();
        const int hbase = h * 256 + w0;
        for (int idx = tid; idx < 64 * 32; idx += 256) {
            int c  = idx >> 5;
            int wq = (idx & 31) * 4;
            float4 v = *(const float4*)(lds_t + c * CST + wq);
            *(float4*)(outF32 + ((long)n * 64 + c) * 65536L + hbase + wq) = v;
        }
#undef CST
    }
}

// =====================================================================
// launch
// =====================================================================
extern "C" void kernel_launch(void* const* d_in, const int* in_sizes, int n_in,
                              void* d_out, int out_size, void* d_ws, size_t ws_size,
                              hipStream_t stream) {
    const float* x  = (const float*)d_in[0];
    const float* w1 = (const float*)d_in[1];
    const float* b1 = (const float*)d_in[2];
    const float* w2 = (const float*)d_in[3];
    const float* b2 = (const float*)d_in[4];
    const float* w3 = (const float*)d_in[5];
    const float* b3 = (const float*)d_in[6];
    const float* pa = (const float*)d_in[7];

    // workspace layout (bf16 stored as unsigned short)
    unsigned short* x0  = (unsigned short*)d_ws;       // M*64
    unsigned short* y   = x0 + MTOT * 64;              // M*64
    unsigned short* w2p = y + MTOT * 64;               // 36864
    unsigned short* w3p = w2p + WP_ELEMS;              // 36864

    pack_w_kernel<<<dim3((WP_ELEMS + 255) / 256), dim3(256), 0, stream>>>(w2, w2p);
    pack_w_kernel<<<dim3((WP_ELEMS + 255) / 256), dim3(256), 0, stream>>>(w3, w3p);

    conv1_kernel<<<dim3((unsigned)(MTOT / 4)), dim3(256), 0, stream>>>(x, w1, b1, pa, x0);

    conv_wmma_kernel<true><<<dim3((unsigned)(MTOT / 128)), dim3(256), 0, stream>>>(
        x0, w2p, b2, x0, pa, y, nullptr);

    conv_wmma_kernel<false><<<dim3((unsigned)(MTOT / 128)), dim3(256), 0, stream>>>(
        y, w3p, b3, nullptr, pa, nullptr, (float*)d_out);
}
